// BoneLinear_51144470561223
// MI455X (gfx1250) — compile-verified, matrix operand order checked by reading, not run on hardware
//
#include <hip/hip_runtime.h>

// BoneLinear on gfx1250 (MI455X), compile-only target.
//
//   out = x @ W^T + pooled(x) @ disha
// folds algebraically into a single GEMM:
//   out = x @ W_eff^T,  W_eff[n,k] = W[n,k] + disha[k & 63, n]
//
// M = 8*4096 = 32768, N = 1024, K = 1024.
// W_eff precomputed to bf16 in d_ws (2 MB). GEMM converts x f32->bf16 during
// global->LDS staging and runs v_wmma_f32_16x16x32_bf16 with f32 accumulators.
//
// Tile config: block 128(M) x 256(N), KT=32, 8 waves of 32 (wave32), each wave
// owns a 64x64 macro-tile = 4x4 WMMA fragments -> 16 wmma per 16 ds_load_b128
// (1.0 LDS loads per WMMA, 4x reuse of both operands). Double-buffered LDS
// (60 KB of 320 KB/WGP), bank-conflict-free padded rows (stride 40 bf16).

typedef __attribute__((ext_vector_type(16))) __bf16 v16bf;
typedef __attribute__((ext_vector_type(8)))  __bf16 bf16x8;
typedef __attribute__((ext_vector_type(4)))  __bf16 bf16x4;
typedef __attribute__((ext_vector_type(8)))  float  v8f;

union V16 { v16bf v; bf16x8 h[2]; };

#define M_DIM 32768
#define N_DIM 1024
#define K_DIM 1024
#define BM    128
#define BN    256
#define KT    32
#define NKT   (K_DIM / KT)   // 32
#define LDSK  40             // padded LDS row stride (bf16 elems): bank-conflict-free

// --- pre-pass: W_eff[n,k] = W[n,k] + disha[k & 63, n], cast to bf16 ----------
__global__ __launch_bounds__(256)
void bone_weff_kernel(const float* __restrict__ w,
                      const float* __restrict__ dsh,
                      __bf16* __restrict__ weff) {
  int idx = blockIdx.x * 256 + threadIdx.x;   // idx = n*1024 + k
  int n = idx >> 10;
  int k = idx & 1023;
  weff[idx] = (__bf16)(w[idx] + dsh[((k & 63) << 10) + n]);
}

// --- main GEMM: out[M,N] = x[M,K] @ W_eff[N,K]^T -----------------------------
__global__ __launch_bounds__(256)
void bone_gemm_kernel(const float* __restrict__ x,
                      const __bf16* __restrict__ weff,
                      float* __restrict__ out) {
  __shared__ __bf16 Ash[2][BM * LDSK];   // 2 * 128*40*2B = 20 KB
  __shared__ __bf16 Bsh[2][BN * LDSK];   // 2 * 256*40*2B = 40 KB

  const int tid  = threadIdx.x;
  const int lane = tid & 31;
  const int wave = tid >> 5;          // 0..7 -> 2 (M) x 4 (N) wave grid
  const int wm   = (wave >> 2) * 64;  // 0, 64
  const int wn   = (wave & 3) * 64;   // 0, 64, 128, 192
  const int bm   = blockIdx.y * BM;
  const int bn   = blockIdx.x * BN;

  v8f acc[4][4] = {};                 // 4 (M) x 4 (N) 16x16 f32 accumulators

  float4 aReg[4];                     // staging regs: A tile 128x32 f32
  bf16x8 bReg[4];                     //               B tile 256x32 bf16

  auto gloadA = [&](int kt) {
    const float* base = x + (size_t)bm * K_DIM + kt * KT;
#pragma unroll
    for (int p = 0; p < 4; ++p) {
      int i = tid + p * 256;          // 1024 float4 chunks, 8 per 32-f32 row
      int row = i >> 3, c4 = i & 7;
      aReg[p] = *(const float4*)(base + (size_t)row * K_DIM + c4 * 4);
    }
  };
  auto gloadB = [&](int kt) {
    const __bf16* base = weff + (size_t)bn * K_DIM + kt * KT;
#pragma unroll
    for (int p = 0; p < 4; ++p) {
      int i = tid + p * 256;          // 1024 bf16x8 chunks, 4 per 32-bf16 row
      int row = i >> 2, c8 = i & 3;
      bReg[p] = *(const bf16x8*)(base + (size_t)row * K_DIM + c8 * 8);
    }
  };
  auto stageStore = [&](int st) {
#pragma unroll
    for (int p = 0; p < 4; ++p) {
      int i = tid + p * 256;
      int row = i >> 3, c4 = i & 7;
      bf16x4 v;
      v[0] = (__bf16)aReg[p].x; v[1] = (__bf16)aReg[p].y;
      v[2] = (__bf16)aReg[p].z; v[3] = (__bf16)aReg[p].w;
      *(bf16x4*)&Ash[st][row * LDSK + c4 * 4] = v;
    }
#pragma unroll
    for (int p = 0; p < 4; ++p) {
      int i = tid + p * 256;
      int row = i >> 2, c8 = i & 3;
      *(bf16x8*)&Bsh[st][row * LDSK + c8 * 8] = bReg[p];
    }
  };
  auto compute = [&](int st) {
    // ISA bf16 A/B fragment layout: lanes 0-15 hold K 0-7 & 16-23,
    // lanes 16-31 hold K 8-15 & 24-31, for row/col = lane & 15.
    const int r  = lane & 15;
    const int k0 = (lane >> 4) * 8;
    V16 bfr[4];                       // B-frags resident, A-frags streamed
#pragma unroll
    for (int nf = 0; nf < 4; ++nf) {
      const __bf16* p = &Bsh[st][(wn + nf * 16 + r) * LDSK + k0];
      bfr[nf].h[0] = *(const bf16x8*)p;
      bfr[nf].h[1] = *(const bf16x8*)(p + 16);
    }
#pragma unroll
    for (int mf = 0; mf < 4; ++mf) {
      V16 af;
      const __bf16* p = &Ash[st][(wm + mf * 16 + r) * LDSK + k0];
      af.h[0] = *(const bf16x8*)p;
      af.h[1] = *(const bf16x8*)(p + 16);
#pragma unroll
      for (int nf = 0; nf < 4; ++nf)
        acc[mf][nf] = __builtin_amdgcn_wmma_f32_16x16x32_bf16(
            false, af.v, false, bfr[nf].v,
            (short)0, acc[mf][nf], false, false);
    }
  };

  // software pipeline over K, double-buffered LDS
  gloadA(0); gloadB(0);
  stageStore(0);
  __syncthreads();
#pragma unroll 1
  for (int kt = 0; kt < NKT; ++kt) {
    const int cur = kt & 1;
    if (kt + 1 < NKT) { gloadA(kt + 1); gloadB(kt + 1); }
    compute(cur);
    if (kt + 1 < NKT) stageStore(cur ^ 1);
    __syncthreads();
  }

  // epilogue: C/D layout -> lane (lane&15 = N), VGPR v -> M = 8*(lane>>4)+v
  const int cn = lane & 15;
  const int cm = (lane >> 4) * 8;
#pragma unroll
  for (int mf = 0; mf < 4; ++mf) {
    const int rbase = bm + wm + mf * 16 + cm;
#pragma unroll
    for (int nf = 0; nf < 4; ++nf) {
      const int col = bn + wn + nf * 16 + cn;
#pragma unroll
      for (int v = 0; v < 8; ++v)
        out[(size_t)(rbase + v) * N_DIM + col] = acc[mf][nf][v];
    }
  }
}

extern "C" void kernel_launch(void* const* d_in, const int* in_sizes, int n_in,
                              void* d_out, int out_size, void* d_ws, size_t ws_size,
                              hipStream_t stream) {
  const float* x     = (const float*)d_in[0];   // [8,4096,1024] f32
  const float* w     = (const float*)d_in[1];   // [1024,1024]   f32
  const float* disha = (const float*)d_in[2];   // [64,1024]     f32
  float* out = (float*)d_out;                   // [8,4096,1024] f32
  __bf16* weff = (__bf16*)d_ws;                 // 1024*1024 bf16 = 2 MB scratch

  // 1) fold Bone adapter into the weight, cast to bf16
  bone_weff_kernel<<<(N_DIM * K_DIM) / 256, 256, 0, stream>>>(w, disha, weff);

  // 2) single bf16-WMMA GEMM, fused f32->bf16 conversion of x in staging
  dim3 grid(N_DIM / BN, M_DIM / BM);   // (4, 256)
  bone_gemm_kernel<<<grid, 256, 0, stream>>>(x, weff, out);
}
